// VNetProto_64561948393828
// MI455X (gfx1250) — matrix operand order repaired and checked by reading, not compiled, not run on hardware
//
#include <hip/hip_runtime.h>
#include <hip/hip_bf16.h>
#include <math.h>

typedef __attribute__((ext_vector_type(16))) __bf16 v16bf;
typedef __attribute__((ext_vector_type(8)))  __bf16 v8bf;
typedef __attribute__((ext_vector_type(8)))  float  v8f;

#define CC      128
#define DHW     110592
#define NTOT    221184
#define LNEPS   1e-5f
#define BNEPS   1e-5f
#define EPSINV  20.0f      /* 1/EPS_SINK */
#define TAUINV  2.0f       /* 1/TAU */
#define MOMC    0.999f
#define RBLK    432        /* 432*512 == NTOT */
#define USTRIDE 1552       /* 12*128 sums + 12 counts, padded */

// ---------------------------------------------------------------- prep ------
__global__ void prep_kernel(const float* __restrict__ w1, const float* __restrict__ w2,
                            const float* __restrict__ c1b, const float* __restrict__ bng,
                            const float* __restrict__ bnb, const float* __restrict__ bnm,
                            const float* __restrict__ bnv,
                            const float* __restrict__ pl, const float* __restrict__ pu,
                            __bf16* __restrict__ w1bf, __bf16* __restrict__ w2bf,
                            float* __restrict__ scale1, float* __restrict__ bias1,
                            float* __restrict__ plN, float* __restrict__ puN) {
  int tid = threadIdx.x;
  for (int i = tid; i < CC * CC; i += blockDim.x) {
    w1bf[i] = (__bf16)w1[i];
    w2bf[i] = (__bf16)w2[i];
  }
  if (tid < CC) {
    float s = bng[tid] * rsqrtf(bnv[tid] + BNEPS);   // fold BN into affine
    scale1[tid] = s;
    bias1[tid] = (c1b[tid] - bnm[tid]) * s + bnb[tid];
  }
  if (tid < 24) {                                    // l2-normalize prototypes
    const float* src = (tid < 12) ? (pl + tid * CC) : (pu + (tid - 12) * CC);
    float* dst = (tid < 12) ? (plN + tid * CC) : (puN + (tid - 12) * CC);
    float ss = 0.f;
    for (int c = 0; c < CC; ++c) ss += src[c] * src[c];
    float inv = 1.f / fmaxf(sqrtf(ss), 1e-12f);
    for (int c = 0; c < CC; ++c) dst[c] = src[c] * inv;
  }
}

// --------------------------------------------------------------- embed ------
// 4 waves/block, 16 voxels/wave. Two bf16 WMMA GEMMs (K=128), then
// l2 -> LN -> l2 per voxel, then 24 prototype dots (masks/logits/out_seg).
__global__ void __launch_bounds__(128)
embed_kernel(const float* __restrict__ x,
             const __bf16* __restrict__ w1bf, const __bf16* __restrict__ w2bf,
             const float* __restrict__ scale1, const float* __restrict__ bias1,
             const float* __restrict__ c2b,
             const float* __restrict__ fg, const float* __restrict__ fb,
             const float* __restrict__ mg, const float* __restrict__ mb,
             const float* __restrict__ plN, const float* __restrict__ puN,
             float* __restrict__ cOut,
             float* __restrict__ outseg_l, float* __restrict__ outseg_u,
             float* __restrict__ logits_l, float* __restrict__ logits_u) {
  __shared__ __align__(32) float sm[4 * 16 * 132];
  const int wave = threadIdx.x >> 5;
  const int lane = threadIdx.x & 31;
  const int group = blockIdx.x * 4 + wave;
  const int n0 = group * 16;
  const int b = n0 / DHW;
  const int s0 = n0 % DHW;
  const float* xb = x + (size_t)b * CC * DHW + s0;
  float* smw = sm + wave * (16 * 132);
  __bf16* h1 = (__bf16*)smw;                 // aliased: bf16 [16 voxels][128 ch]

  const int col = lane & 15;                 // voxel within group / A row (M)
  const int hsel = lane >> 4;                // lane half
  const int ah = hsel * 16;                  // A K-offset: contiguous 16 K per lane half (matches B)
  const int bh = hsel * 16;                  // B K-offset: contiguous 16 K per lane half (matches A)

  v8f acc[8];
#pragma unroll
  for (int i = 0; i < 8; ++i)
#pragma unroll
    for (int j = 0; j < 8; ++j) acc[i][j] = 0.f;

  // ---- GEMM1: h = W1 @ x  (8 o-tiles x 4 k-steps of 16x16x32 bf16 WMMA)
#pragma unroll
  for (int kk = 0; kk < 4; ++kk) {
    union { v16bf v; __bf16 e[16]; } bu;
    const float* xc = xb + (size_t)(kk * 32 + bh) * DHW + col;
    if (kk < 3) __builtin_prefetch(xb + (size_t)((kk + 1) * 32 + bh) * DHW + col, 0, 1);
#pragma unroll
    for (int i = 0; i < 16; ++i) bu.e[i] = (__bf16)xc[(size_t)i * DHW];
#pragma unroll
    for (int ot = 0; ot < 8; ++ot) {
      const __bf16* wr = w1bf + (ot * 16 + col) * CC + kk * 32 + ah;
      union { v16bf v; v8bf h[2]; } au;
      au.h[0] = *(const v8bf*)wr;
      au.h[1] = *(const v8bf*)(wr + 8);
      acc[ot] = __builtin_amdgcn_wmma_f32_16x16x32_bf16(
          false, au.v, false, bu.v, (short)0, acc[ot], false, false);
    }
  }

  // ---- fused BN + ReLU; stash h1 to LDS as [voxel][channel] bf16
#pragma unroll
  for (int ot = 0; ot < 8; ++ot)
#pragma unroll
    for (int j = 0; j < 8; ++j) {
      int o = ot * 16 + j + hsel * 8;        // C-matrix layout: VGPR j, lane half
      float v = acc[ot][j] * scale1[o] + bias1[o];
      h1[col * CC + o] = (__bf16)fmaxf(v, 0.f);
    }
  __builtin_amdgcn_wave_barrier();

  // ---- GEMM2: h2 = W2 @ h1
  v8f acc2[8];
#pragma unroll
  for (int i = 0; i < 8; ++i)
#pragma unroll
    for (int j = 0; j < 8; ++j) acc2[i][j] = 0.f;
#pragma unroll
  for (int kk = 0; kk < 4; ++kk) {
    v16bf bv = *(const v16bf*)(h1 + col * CC + kk * 32 + bh);  // 16 contiguous bf16
#pragma unroll
    for (int ot = 0; ot < 8; ++ot) {
      const __bf16* wr = w2bf + (ot * 16 + col) * CC + kk * 32 + ah;
      union { v16bf v; v8bf h[2]; } au;
      au.h[0] = *(const v8bf*)wr;
      au.h[1] = *(const v8bf*)(wr + 8);
      acc2[ot] = __builtin_amdgcn_wmma_f32_16x16x32_bf16(
          false, au.v, false, bv, (short)0, acc2[ot], false, false);
    }
  }
  __builtin_amdgcn_wave_barrier();

  // ---- + conv2 bias -> LDS f32 [voxel][132]
#pragma unroll
  for (int ot = 0; ot < 8; ++ot)
#pragma unroll
    for (int j = 0; j < 8; ++j) {
      int o = ot * 16 + j + hsel * 8;
      smw[col * 132 + o] = acc2[ot][j] + c2b[o];
    }
  __builtin_amdgcn_wave_barrier();

  // ---- per-voxel: l2(channel) -> LN -> l2 ; 2 lanes per voxel
  {
    const int v = lane >> 1;
    const int hf = lane & 1;
    float* hv = smw + v * 132;
    const int cb = hf * 64;
    float sv = 0.f, ss = 0.f;
    for (int c = 0; c < 64; ++c) { float f = hv[cb + c]; sv += f; ss += f * f; }
    sv += __shfl_xor(sv, 1, 32);
    ss += __shfl_xor(ss, 1, 32);
    float inv = 1.f / fmaxf(sqrtf(ss), 1e-12f);
    float mu = inv * sv * (1.f / 128.f);
    float ex2 = inv * inv * ss * (1.f / 128.f);
    float rstd = rsqrtf(fmaxf(ex2 - mu * mu, 0.f) + LNEPS);
    float s2 = 0.f;
    for (int c = 0; c < 64; ++c) {
      int ch = cb + c;
      float t = (hv[ch] * inv - mu) * rstd * fg[ch] + fb[ch];
      hv[ch] = t; s2 += t * t;
    }
    s2 += __shfl_xor(s2, 1, 32);
    float inv2 = 1.f / fmaxf(sqrtf(s2), 1e-12f);
    float* crow = cOut + (size_t)(n0 + v) * CC + cb;
    for (int c = 0; c < 64; ++c) {
      float val = hv[cb + c] * inv2;
      hv[cb + c] = val;
      crow[c] = val;
    }
  }
  __builtin_amdgcn_wave_barrier();

  // ---- masks (== logits) + out_seg for both branches (lane halves)
  {
    const int v2 = lane & 15;
    const int br = lane >> 4;
    const int n = n0 + v2;
    const float* P = br ? puN : plN;
    float* lgt = (br ? logits_u : logits_l) + (size_t)n * 12;
    float* oseg = br ? outseg_u : outseg_l;
    const float* crow = smw + v2 * 132;
    float dots[12];
#pragma unroll
    for (int p = 0; p < 12; ++p) dots[p] = 0.f;
    for (int c = 0; c < CC; ++c) {
      float f = crow[c];
#pragma unroll
      for (int p = 0; p < 12; ++p) dots[p] += f * P[p * CC + c];
    }
#pragma unroll
    for (int p = 0; p < 12; ++p) lgt[p] = dots[p];
    float mx[3];
#pragma unroll
    for (int k = 0; k < 3; ++k)
      mx[k] = fmaxf(fmaxf(dots[k * 4], dots[k * 4 + 1]),
                    fmaxf(dots[k * 4 + 2], dots[k * 4 + 3]));
    float mu3 = (mx[0] + mx[1] + mx[2]) * (1.f / 3.f);
    float var3 = ((mx[0] - mu3) * (mx[0] - mu3) + (mx[1] - mu3) * (mx[1] - mu3) +
                  (mx[2] - mu3) * (mx[2] - mu3)) * (1.f / 3.f);
    float r3 = rsqrtf(var3 + LNEPS);
    int s = s0 + v2;
#pragma unroll
    for (int k = 0; k < 3; ++k)
      oseg[(size_t)b * 3 * DHW + (size_t)k * DHW + s] = (mx[k] - mu3) * r3 * mg[k] + mb[k];
  }
}

// -------------------------------------------------- sinkhorn reductions -----
// Rank-1 form: L[k,m,n] = E[n,m,k] * a[k,m] * b[k,n], with
// b = 1/sum_m(E * a * Bn[k]) (safe). phase 0: sum E + Bn counts.
// phase t>=1: sum_n E*b_{t-1}. Fixed-order block reduction (deterministic).
__global__ void __launch_bounds__(256)
sink_reduce(const float* __restrict__ masks12, const int* __restrict__ gt,
            const float* __restrict__ a, const float* __restrict__ Bn,
            float* __restrict__ partials, int phase) {
  __shared__ float red[256];
  const int tid = threadIdx.x;
  float loc[16];
#pragma unroll
  for (int i = 0; i < 16; ++i) loc[i] = 0.f;
  for (int it = 0; it < 2; ++it) {
    int n = blockIdx.x * 512 + tid + it * 256;
    int k = gt[n];
    const float* mr = masks12 + (size_t)n * 12;
    float E[4];
#pragma unroll
    for (int m = 0; m < 4; ++m) E[m] = expf(mr[k * 4 + m] * EPSINV);
    float bcoef = 1.f;
    if (phase >= 2) {
      float s = 0.f;
#pragma unroll
      for (int m = 0; m < 4; ++m) s += E[m] * a[k * 4 + m];
      s *= Bn[k];
      bcoef = (s > 0.f) ? 1.f / s : 1.f;
    }
#pragma unroll
    for (int kk = 0; kk < 3; ++kk) {
      float match = (k == kk) ? 1.f : 0.f;
#pragma unroll
      for (int m = 0; m < 4; ++m) loc[kk * 4 + m] += match * E[m] * bcoef;
      if (phase == 0) loc[12 + kk] += match;
    }
  }
  for (int comp = 0; comp < 16; ++comp) {
    red[tid] = loc[comp];
    __syncthreads();
    for (int st = 128; st > 0; st >>= 1) {
      if (tid < st) red[tid] += red[tid + st];
      __syncthreads();
    }
    if (tid == 0) partials[blockIdx.x * 16 + comp] = red[0];
    __syncthreads();
  }
}

__global__ void sink_finish(const float* __restrict__ partials, int nblk,
                            float* __restrict__ a, float* __restrict__ Bn, int phase) {
  __shared__ float tot[16];
  int tid = threadIdx.x;
  if (tid < 16) {
    float s = 0.f;
    for (int b2 = 0; b2 < nblk; ++b2) s += partials[b2 * 16 + tid];
    tot[tid] = s;
  }
  __syncthreads();
  if (tid == 0) {
    if (phase == 0) {
      for (int k = 0; k < 3; ++k) {
        Bn[k] = tot[12 + k];
        float S = tot[k * 4] + tot[k * 4 + 1] + tot[k * 4 + 2] + tot[k * 4 + 3];
        float ai = 1.f / ((S == 0.f) ? 1.f : S);
        for (int m = 0; m < 4; ++m) a[k * 4 + m] = ai;
      }
    } else {
      for (int k = 0; k < 3; ++k) {
        float bnd = (Bn[k] == 0.f) ? 1.f : Bn[k];
        for (int m = 0; m < 4; ++m) {
          int i = k * 4 + m;
          float R = a[i] * tot[i];                       // rowsum
          float ah = a[i] / (((R == 0.f) ? 1.f : R) * 4.f);
          a[i] = ah / bnd;
        }
      }
    }
  }
}

// -------------------------------- assignment, targets, proto partials -------
__global__ void __launch_bounds__(256)
assign_kernel(const float* __restrict__ masks12, const int* __restrict__ gt,
              const float* __restrict__ gu, const float* __restrict__ a,
              const float* __restrict__ Bn, const float* __restrict__ cMat,
              const float* __restrict__ mg, const float* __restrict__ mb,
              float* __restrict__ target, float* __restrict__ upd) {
  __shared__ short bucket[512];
  const int tid = threadIdx.x;
  const int base = blockIdx.x * 512;
  for (int it = 0; it < 2; ++it) {
    int lv = tid + it * 256;
    int n = base + lv;
    const float* mr = masks12 + (size_t)n * 12;
    float d[12];
#pragma unroll
    for (int p = 0; p < 12; ++p) d[p] = mr[p];
    float mx[3];
#pragma unroll
    for (int k = 0; k < 3; ++k)
      mx[k] = fmaxf(fmaxf(d[k * 4], d[k * 4 + 1]), fmaxf(d[k * 4 + 2], d[k * 4 + 3]));
    float mu3 = (mx[0] + mx[1] + mx[2]) * (1.f / 3.f);
    float var3 = ((mx[0] - mu3) * (mx[0] - mu3) + (mx[1] - mu3) * (mx[1] - mu3) +
                  (mx[2] - mu3) * (mx[2] - mu3)) * (1.f / 3.f);
    float r3 = rsqrtf(var3 + LNEPS);
    int pred = 0; float best = -INFINITY;
#pragma unroll
    for (int k = 0; k < 3; ++k) {
      float o = (mx[k] - mu3) * r3 * mg[k] + mb[k];
      if (o > best) { best = o; pred = k; }              // first-max like jnp.argmax
    }
    int kg = gt[n];
    float w[4], S = 0.f;
#pragma unroll
    for (int m = 0; m < 4; ++m) {
      float E = expf(d[kg * 4 + m] * EPSINV);
      w[m] = E * a[kg * 4 + m] * Bn[kg];
      S += w[m];
    }
    float Lv[4];
#pragma unroll
    for (int m = 0; m < 4; ++m) Lv[m] = (S > 0.f) ? w[m] / S : 0.f;
    int idx = 0; float bl = Lv[0];
#pragma unroll
    for (int m = 1; m < 4; ++m) if (Lv[m] > bl) { bl = Lv[m]; idx = m; }
    target[n] = (float)(idx + 4 * kg);
    short bk = -1;
    if (pred == kg) {                                    // correct prediction
      const float* ur = gu + ((size_t)kg * NTOT + n) * 4;
      int ms = 0; float yb = -INFINITY;
#pragma unroll
      for (int m = 0; m < 4; ++m) {
        float u = fminf(fmaxf(ur[m], 1e-10f), 1.0f);
        float g = -logf(-logf(u));
        float y = (Lv[m] + g) * TAUINV;                  // argmax(softmax)==argmax(logits)
        if (y > yb) { yb = y; ms = m; }
      }
      bk = (short)(kg * 4 + ms);
    }
    bucket[lv] = bk;
  }
  __syncthreads();
  // deterministic bucket accumulation: slot (r,c) scans voxels in fixed order
#pragma unroll
  for (int j = 0; j < 6; ++j) {
    int slot = tid + j * 256;
    int r = slot >> 7, c = slot & 127;
    float s = 0.f; int cnt = 0;
    for (int lv = 0; lv < 512; ++lv)
      if (bucket[lv] == r) { s += cMat[(size_t)(base + lv) * CC + c]; cnt++; }
    upd[(size_t)blockIdx.x * USTRIDE + slot] = s;
    if (c == 0) upd[(size_t)blockIdx.x * USTRIDE + 1536 + r] = (float)cnt;
  }
}

__global__ void __launch_bounds__(128)
proto_finish(const float* __restrict__ upd, int nblk,
             const float* __restrict__ pN, float* __restrict__ outP) {
  __shared__ float rtmp[128];
  __shared__ float cshare;
  int c = threadIdx.x;
  for (int r = 0; r < 12; ++r) {
    float s = 0.f;
    for (int b2 = 0; b2 < nblk; ++b2) s += upd[(size_t)b2 * USTRIDE + r * 128 + c];
    if (c == 0) {
      float cs = 0.f;
      for (int b2 = 0; b2 < nblk; ++b2) cs += upd[(size_t)b2 * USTRIDE + 1536 + r];
      cshare = cs;
    }
    rtmp[c] = s * s;
    __syncthreads();
    float cnt = cshare;
    for (int st = 64; st > 0; st >>= 1) { if (c < st) rtmp[c] += rtmp[c + st]; __syncthreads(); }
    float nf = rtmp[0];
    __syncthreads();
    float fn = s / fmaxf(sqrtf(nf), 1e-12f);
    float p = pN[r * 128 + c];
    float np = (cnt > 0.f) ? (MOMC * p + (1.f - MOMC) * fn) : p;
    rtmp[c] = np * np;
    __syncthreads();
    for (int st = 64; st > 0; st >>= 1) { if (c < st) rtmp[c] += rtmp[c + st]; __syncthreads(); }
    float n2 = rtmp[0];
    __syncthreads();
    outP[r * 128 + c] = np / fmaxf(sqrtf(n2), 1e-12f);
    __syncthreads();
  }
}

// ---------------------------------------------------------------- host ------
extern "C" void kernel_launch(void* const* d_in, const int* in_sizes, int n_in,
                              void* d_out, int out_size, void* d_ws, size_t ws_size,
                              hipStream_t stream) {
  (void)in_sizes; (void)n_in; (void)out_size; (void)ws_size;
  const float* x   = (const float*)d_in[0];
  const float* w1  = (const float*)d_in[1];
  const float* c1b = (const float*)d_in[2];
  const float* bng = (const float*)d_in[3];
  const float* bnb = (const float*)d_in[4];
  const float* bnm = (const float*)d_in[5];
  const float* bnv = (const float*)d_in[6];
  const float* w2  = (const float*)d_in[7];
  const float* c2b = (const float*)d_in[8];
  const float* fg  = (const float*)d_in[9];
  const float* fb  = (const float*)d_in[10];
  const float* mg  = (const float*)d_in[11];
  const float* mb  = (const float*)d_in[12];
  const float* pl  = (const float*)d_in[13];
  const float* pu  = (const float*)d_in[14];
  const float* gu  = (const float*)d_in[15];
  const int*   gts = (const int*)d_in[16];
  const int*   pss = (const int*)d_in[17];

  float* out = (float*)d_out;
  float* outseg_l = out;
  float* logits_l = outseg_l + (size_t)2 * 3 * DHW;
  float* target_l = logits_l + (size_t)NTOT * 12;
  float* outseg_u = target_l + NTOT;
  float* logits_u = outseg_u + (size_t)2 * 3 * DHW;
  float* target_u = logits_u + (size_t)NTOT * 12;
  float* new_pl   = target_u + NTOT;
  float* new_pu   = new_pl + 1536;

  float* wsf = (float*)d_ws;
  size_t off = 0;
  float*  cMat  = wsf;                 off += (size_t)NTOT * CC;
  __bf16* w1bf  = (__bf16*)(wsf + off); off += 8192;
  __bf16* w2bf  = (__bf16*)(wsf + off); off += 8192;
  float*  scl1  = wsf + off;           off += 128;
  float*  bia1  = wsf + off;           off += 128;
  float*  plN   = wsf + off;           off += 1536;
  float*  puN   = wsf + off;           off += 1536;
  float*  red   = wsf + off;           off += (size_t)RBLK * 16;
  float*  aS    = wsf + off;           off += 16;
  float*  BnS   = wsf + off;           off += 16;
  float*  upd   = wsf + off;           off += (size_t)RBLK * USTRIDE;

  prep_kernel<<<1, 256, 0, stream>>>(w1, w2, c1b, bng, bnb, bnm, bnv, pl, pu,
                                     w1bf, w2bf, scl1, bia1, plN, puN);
  embed_kernel<<<NTOT / 64, 128, 0, stream>>>(x, w1bf, w2bf, scl1, bia1, c2b,
                                              fg, fb, mg, mb, plN, puN, cMat,
                                              outseg_l, outseg_u, logits_l, logits_u);

  for (int br = 0; br < 2; ++br) {
    const float* masks = br ? logits_u : logits_l;
    const int*   gt    = br ? pss : gts;
    const float* gub   = gu + (size_t)br * 3 * NTOT * 4;
    const float* P     = br ? puN : plN;
    float* tgt         = br ? target_u : target_l;
    float* npo         = br ? new_pu : new_pl;

    sink_reduce<<<RBLK, 256, 0, stream>>>(masks, gt, aS, BnS, red, 0);
    sink_finish<<<1, 32, 0, stream>>>(red, RBLK, aS, BnS, 0);
    for (int t = 1; t <= 3; ++t) {
      sink_reduce<<<RBLK, 256, 0, stream>>>(masks, gt, aS, BnS, red, t);
      sink_finish<<<1, 32, 0, stream>>>(red, RBLK, aS, BnS, t);
    }
    assign_kernel<<<RBLK, 256, 0, stream>>>(masks, gt, gub, aS, BnS, cMat, mg, mb, tgt, upd);
    proto_finish<<<1, 128, 0, stream>>>(upd, RBLK, P, npo);
  }
}
